// MP_28595892257560
// MI455X (gfx1250) — matrix operand order repaired and checked by reading, not compiled, not run on hardware
//
#include <hip/hip_runtime.h>
#include <hip/hip_bf16.h>
#include <math.h>

// ---------------------------------------------------------------------------
// Problem constants (from reference)
// ---------------------------------------------------------------------------
#define NN    50000
#define PP    3
#define EE    800000
#define FEAT  1000
#define KPAD  1024
#define HH    256
#define PH    512
#define BN_EPS 1e-5f
#define MTILES 3125            // NN / 16

typedef __attribute__((ext_vector_type(16))) __bf16 v16bf;
typedef __attribute__((ext_vector_type(8)))  __bf16 v8bf;
typedef __attribute__((ext_vector_type(8)))  float  v8f;

// ---------------------------------------------------------------------------
// Prep kernels
// ---------------------------------------------------------------------------

// feats [N x FEAT] fp32 -> bf16 [N x KPAD], zero padded (row-major A operand)
__global__ void cvt_pad_feats(const float* __restrict__ f, __bf16* __restrict__ o) {
    long i = (long)blockIdx.x * blockDim.x + threadIdx.x;
    if (i >= (long)NN * KPAD) return;
    int n = (int)(i >> 10);         // /KPAD
    int k = (int)(i & (KPAD - 1));
    float v = (k < FEAT) ? f[(long)n * FEAT + k] : 0.f;
    o[i] = (__bf16)v;
}

// W [O x I] fp32 row-major -> swizzled bf16 B operand (B = W^T, [Kp x O], zero pad)
// Fragment-contiguous layout: o[(((kt*NT + nt)*32 + lane)*16 + i]
//   = B[kt*32 + (lane>>4)*16 + i][nt*16 + (lane&15)]
// so each lane's 16x16x32-bf16 B fragment is one contiguous 32-byte block.
__global__ void prep_w_sw(const float* __restrict__ w, __bf16* __restrict__ o,
                          int O, int I, int Kp) {
    long idx = (long)blockIdx.x * blockDim.x + threadIdx.x;
    if (idx >= (long)Kp * O) return;
    int i    = (int)(idx & 15);
    int lane = (int)((idx >> 4) & 31);
    long t   = idx >> 9;
    int NT = O >> 4;
    int nt = (int)(t % NT);
    int kt = (int)(t / NT);
    int n = nt * 16 + (lane & 15);
    int k = kt * 32 + (lane >> 4) * 16 + i;
    float v = (k < I) ? w[(long)n * I + k] : 0.f;
    o[idx] = (__bf16)v;
}

// out[r][c] = bias[c]  (accumulator seed for SpMM)
__global__ void init_bias_rows(float* __restrict__ o, const float* __restrict__ bias,
                               long total, int cols) {
    long i = (long)blockIdx.x * blockDim.x + threadIdx.x;
    if (i >= total) return;
    o[i] = bias[i & (cols - 1)];
}

// ---------------------------------------------------------------------------
// bf16 WMMA GEMM: 256 threads = 8 waves; each wave computes a 16x32 strip
// (two 16x16 n-tiles sharing one A fragment), K unrolled x2 -> 4 WMMA / iter.
//   act: 0 = none, 1 = ELU
//   colsum != nullptr : tanh(val+bias) column-sum epilogue (attention), no store
//   stats  != nullptr : accumulate per-column sum / sumsq (BatchNorm stats)
// ---------------------------------------------------------------------------
__global__ void wmma_gemm_bf16(const __bf16* __restrict__ A, const __bf16* __restrict__ Bsw,
                               const float* __restrict__ bias,
                               float* __restrict__ Cf, __bf16* __restrict__ Cb,
                               float* __restrict__ stats, float* __restrict__ colsum,
                               int Ncols, int K, int act, int Mtiles) {
    const int wid  = threadIdx.x >> 5;
    const int lane = threadIdx.x & 31;
    const int mtile = blockIdx.y * 8 + wid;
    if (mtile >= Mtiles) return;
    const int np = blockIdx.x;             // column-pair index (32 cols)
    const int m0 = mtile * 16;
    const int NT = Ncols >> 4;
    const int hi = lane >> 4;
    const int mrow = lane & 15;

    v8f acc[2] = {{}, {}};
    const __bf16* Ap  = A + (long)(m0 + mrow) * K;
    const long bstep = (long)NT * 512;     // elements per kt block (NT*32*16)
    const __bf16* Bp0 = Bsw + ((long)(2 * np) * 32 + lane) * 16;
    const __bf16* Bp1 = Bp0 + 512;

    const int KT = K >> 5;
    for (int kt = 0; kt < KT; kt += 2) {
        #pragma unroll
        for (int u = 0; u < 2; ++u) {
            const int k0 = (kt + u) << 5;
            // A fragment: two contiguous 16B chunks -> global_load_b128 x2
            v8bf a0 = *reinterpret_cast<const v8bf*>(Ap + k0 + hi * 16);
            v8bf a1 = *reinterpret_cast<const v8bf*>(Ap + k0 + hi * 16 + 8);
            v16bf a = __builtin_shufflevector(a0, a1,
                        0, 1, 2, 3, 4, 5, 6, 7, 8, 9, 10, 11, 12, 13, 14, 15);
            // B fragments: contiguous 32B per lane -> global_load_b128 x2 each
            v16bf b0 = *reinterpret_cast<const v16bf*>(Bp0 + (long)(kt + u) * bstep);
            v16bf b1 = *reinterpret_cast<const v16bf*>(Bp1 + (long)(kt + u) * bstep);
            acc[0] = __builtin_amdgcn_wmma_f32_16x16x32_bf16(
                        false, a, false, b0, (short)0, acc[0], false, false);
            acc[1] = __builtin_amdgcn_wmma_f32_16x16x32_bf16(
                        false, a, false, b1, (short)0, acc[1], false, false);
        }
        __builtin_prefetch(Ap + ((kt + 2) << 5), 0, 0);   // global_prefetch_b8
    }

    #pragma unroll
    for (int t = 0; t < 2; ++t) {
        const int n = np * 32 + t * 16 + (lane & 15);
        const float bval = bias ? bias[n] : 0.f;

        if (colsum) {   // attention epilogue: sum_m tanh(x + b) per column
            float s = 0.f;
            #pragma unroll
            for (int v = 0; v < 8; ++v) s += tanhf(acc[t][v] + bval);
            s += __shfl_xor(s, 16);
            if (lane < 16) atomicAdd(&colsum[n], s);
            continue;
        }

        float lsum = 0.f, lsq = 0.f;
        #pragma unroll
        for (int v = 0; v < 8; ++v) {
            float val = acc[t][v] + bval;
            if (act == 1) val = (val > 0.f) ? val : expm1f(val);   // ELU
            const int m = m0 + v + hi * 8;                         // C/D layout
            const long idx = (long)m * Ncols + n;
            if (Cf) Cf[idx] = val;
            if (Cb) Cb[idx] = (__bf16)val;
            lsum += val; lsq += val * val;
        }
        if (stats) {
            lsum += __shfl_xor(lsum, 16);
            lsq  += __shfl_xor(lsq, 16);
            if (lane < 16) {
                atomicAdd(&stats[n], lsum);
                atomicAdd(&stats[Ncols + n], lsq);
            }
        }
    }
}

// ---------------------------------------------------------------------------
// SpMM scatter:  acc[dst] += vals * ft[src]   (64 threads/edge, float4 lanes)
// ---------------------------------------------------------------------------
__global__ void spmm_scatter(const int* __restrict__ src, const int* __restrict__ dst,
                             const float* __restrict__ vals,
                             const float* __restrict__ ft, float* __restrict__ acc) {
    long t = (long)blockIdx.x * blockDim.x + threadIdx.x;
    int e = (int)(t >> 6);
    if (e >= EE) return;
    int c = (int)(t & 63) * 4;
    int s = src[e], d = dst[e];
    float v = vals[e];
    const float4 f = *reinterpret_cast<const float4*>(ft + (long)s * HH + c);
    float* o = acc + (long)d * HH + c;
    atomicAdd(o + 0, v * f.x);
    atomicAdd(o + 1, v * f.y);
    atomicAdd(o + 2, v * f.z);
    atomicAdd(o + 3, v * f.w);
}

// PReLU in-place on fp32 accumulator + emit bf16 copy
__global__ void prelu_store(float* __restrict__ v, __bf16* __restrict__ vb,
                            const float* __restrict__ a_ptr, long total) {
    long i = (long)blockIdx.x * blockDim.x + threadIdx.x;
    if (i >= total) return;
    float a = a_ptr[0];
    float x = v[i];
    float r = (x >= 0.f) ? x : a * x;
    v[i] = r;
    vb[i] = (__bf16)r;
}

// BatchNorm (training stats) + PReLU, in-place on bf16 x
__global__ void bn_prelu(__bf16* __restrict__ x, const float* __restrict__ stats,
                         const float* __restrict__ g, const float* __restrict__ b,
                         const float* __restrict__ a_ptr, long total) {
    long i = (long)blockIdx.x * blockDim.x + threadIdx.x;
    if (i >= total) return;
    int col = (int)(i & (PH - 1));
    const float inv_n = 1.f / (float)NN;
    float mu  = stats[col] * inv_n;
    float var = stats[PH + col] * inv_n - mu * mu;
    float rs  = rsqrtf(var + BN_EPS);
    float val = ((float)x[i] - mu) * rs * g[col] + b[col];
    float a = a_ptr[0];
    val = (val >= 0.f) ? val : a * val;
    x[i] = (__bf16)val;
}

// ---------------------------------------------------------------------------
// Cosine distillation loss: one wave per row, 5 dot products, shfl reduce
// ---------------------------------------------------------------------------
__global__ void loss_kernel(const float* __restrict__ vp, const float* __restrict__ t1,
                            const float* __restrict__ t2, float* __restrict__ loss) {
    int warp = threadIdx.x >> 5;
    int lane = threadIdx.x & 31;
    int row = blockIdx.x * 8 + warp;
    if (row >= NN) return;
    long base = (long)row * HH + lane * 8;
    float pp = 0, p1 = 0, n1 = 0, p2 = 0, n2 = 0;
    #pragma unroll
    for (int j = 0; j < 8; j += 4) {
        float4 p = *reinterpret_cast<const float4*>(vp + base + j);
        float4 a = *reinterpret_cast<const float4*>(t1 + base + j);
        float4 b = *reinterpret_cast<const float4*>(t2 + base + j);
        pp += p.x*p.x + p.y*p.y + p.z*p.z + p.w*p.w;
        p1 += p.x*a.x + p.y*a.y + p.z*a.z + p.w*a.w;
        n1 += a.x*a.x + a.y*a.y + a.z*a.z + a.w*a.w;
        p2 += p.x*b.x + p.y*b.y + p.z*b.z + p.w*b.w;
        n2 += b.x*b.x + b.y*b.y + b.z*b.z + b.w*b.w;
    }
    #pragma unroll
    for (int m = 16; m; m >>= 1) {
        pp += __shfl_xor(pp, m); p1 += __shfl_xor(p1, m); n1 += __shfl_xor(n1, m);
        p2 += __shfl_xor(p2, m); n2 += __shfl_xor(n2, m);
    }
    if (lane == 0) {
        float c1 = p1 * rsqrtf(fmaxf(pp * n1, 1e-24f));
        float c2 = p2 * rsqrtf(fmaxf(pp * n2, 1e-24f));
        // ((2-2c1) + (2-2c2)) / 2 / N
        atomicAdd(loss, (2.f - c1 - c2) * (1.f / (float)NN));
    }
}

// softmax over beta_p = att . mean_rows(tanh(z W_att^T + b))   (1 wave)
__global__ void beta_kernel(const float* __restrict__ csum, const float* __restrict__ att,
                            float* __restrict__ beta) {
    int lane = threadIdx.x;
    float s[3];
    #pragma unroll
    for (int z = 0; z < 3; ++z) {
        float p = 0.f;
        for (int j = lane; j < HH; j += 32) p += att[j] * csum[z * HH + j];
        #pragma unroll
        for (int m = 16; m; m >>= 1) p += __shfl_xor(p, m);
        s[z] = p * (1.f / (float)NN);
    }
    if (lane == 0) {
        float mx = fmaxf(s[0], fmaxf(s[1], s[2]));
        float e0 = expf(s[0] - mx), e1 = expf(s[1] - mx), e2 = expf(s[2] - mx);
        float inv = 1.f / (e0 + e1 + e2);
        beta[0] = e0 * inv; beta[1] = e1 * inv; beta[2] = e2 * inv;
    }
}

// z_out = beta0*t1 + beta1*t2 + beta2*student
__global__ void combine_kernel(const float* __restrict__ t1, const float* __restrict__ t2,
                               const float* __restrict__ stu, const float* __restrict__ beta,
                               float* __restrict__ out, long total) {
    long i = (long)blockIdx.x * blockDim.x + threadIdx.x;
    if (i >= total) return;
    out[i] = beta[0] * t1[i] + beta[1] * t2[i] + beta[2] * stu[i];
}

// ---------------------------------------------------------------------------
// Host launcher
// ---------------------------------------------------------------------------
extern "C" void kernel_launch(void* const* d_in, const int* in_sizes, int n_in,
                              void* d_out, int out_size, void* d_ws, size_t ws_size,
                              hipStream_t stream) {
    // Inputs (setup_inputs order)
    const float* feats   = (const float*)d_in[0];
    const int*   src     = (const int*)  d_in[1];   // [P,E]
    const int*   dst     = (const int*)  d_in[2];
    const float* vals    = (const float*)d_in[3];
    const float* W_trans = (const float*)d_in[4];
    const float* b_trans = (const float*)d_in[5];
    const float* W_gcn   = (const float*)d_in[6];   // [P,H,H]
    const float* b_gcn   = (const float*)d_in[7];
    const float* a_gcn   = (const float*)d_in[8];
    const float* W1      = (const float*)d_in[9];
    const float* b1      = (const float*)d_in[10];
    const float* bn_g    = (const float*)d_in[11];
    const float* bn_b    = (const float*)d_in[12];
    const float* a_pred  = (const float*)d_in[13];
    const float* W2      = (const float*)d_in[14];
    const float* b2      = (const float*)d_in[15];
    const float* W_att   = (const float*)d_in[16];
    const float* b_att   = (const float*)d_in[17];
    const float* att     = (const float*)d_in[18];

    // Workspace carve-out
    char* wp = (char*)d_ws;
    auto carve = [&](size_t bytes) -> void* {
        void* p = (void*)wp; wp += (bytes + 255) & ~(size_t)255; return p;
    };
    __bf16* featsB = (__bf16*)carve((size_t)NN * KPAD * 2);
    __bf16* WtB    = (__bf16*)carve((size_t)KPAD * HH * 2);
    __bf16* WgB    = (__bf16*)carve((size_t)PP * HH * HH * 2);
    __bf16* W1B    = (__bf16*)carve((size_t)HH * PH * 2);
    __bf16* W2B    = (__bf16*)carve((size_t)PH * HH * 2);
    __bf16* WattB  = (__bf16*)carve((size_t)HH * HH * 2);
    __bf16* hB     = (__bf16*)carve((size_t)NN * HH * 2);
    float*  ft     = (float*) carve((size_t)NN * HH * 4);
    float*  vout[3]; __bf16* vB[3];
    for (int p = 0; p < 3; ++p) vout[p] = (float*)carve((size_t)NN * HH * 4);
    for (int p = 0; p < 3; ++p) vB[p]   = (__bf16*)carve((size_t)NN * HH * 2);
    __bf16* xbuf  = (__bf16*)carve((size_t)NN * PH * 2);
    float*  vpred = (float*) carve((size_t)NN * HH * 4);
    float*  stats = (float*) carve((size_t)2 * PH * 4);
    float*  csum  = (float*) carve((size_t)3 * HH * 4);
    float*  beta  = (float*) carve(256);

    float* zout  = (float*)d_out;
    float* lossp = (float*)d_out + (size_t)NN * HH;

    const long NH  = (long)NN * HH;
    const long NPH = (long)NN * PH;
    const int gy = (MTILES + 7) / 8;   // 8 waves (row tiles) per block

    // Zero accumulators each call (ws/d_out are poisoned, not re-zeroed)
    hipMemsetAsync(stats, 0, 2 * PH * 4, stream);
    hipMemsetAsync(csum, 0, 3 * HH * 4, stream);
    hipMemsetAsync(lossp, 0, 4, stream);

    // ---- prep: bf16 conversions / fragment-swizzled transposes ----
    cvt_pad_feats<<<(int)(((long)NN * KPAD + 255) / 256), 256, 0, stream>>>(feats, featsB);
    prep_w_sw<<<(KPAD * HH + 255) / 256, 256, 0, stream>>>(W_trans, WtB, HH, FEAT, KPAD);
    for (int p = 0; p < 3; ++p)
        prep_w_sw<<<(HH * HH + 255) / 256, 256, 0, stream>>>(W_gcn + (long)p * HH * HH,
                                                             WgB + (long)p * HH * HH, HH, HH, HH);
    prep_w_sw<<<(HH * PH + 255) / 256, 256, 0, stream>>>(W1, W1B, PH, HH, HH);
    prep_w_sw<<<(PH * HH + 255) / 256, 256, 0, stream>>>(W2, W2B, HH, PH, PH);
    prep_w_sw<<<(HH * HH + 255) / 256, 256, 0, stream>>>(W_att, WattB, HH, HH, HH);

    // ---- stage A: h = elu(feats @ W_trans^T + b_trans), bf16 out ----
    wmma_gemm_bf16<<<dim3(HH / 32, gy), 256, 0, stream>>>(
        featsB, WtB, b_trans, nullptr, hB, nullptr, nullptr, HH, KPAD, 1, MTILES);

    // ---- per-view GCN: ft = h @ W^T ; v = prelu(spmm(ft) + b) ----
    for (int p = 0; p < 3; ++p) {
        wmma_gemm_bf16<<<dim3(HH / 32, gy), 256, 0, stream>>>(
            hB, WgB + (long)p * HH * HH, nullptr, ft, nullptr, nullptr, nullptr,
            HH, HH, 0, MTILES);
        init_bias_rows<<<(int)((NH + 255) / 256), 256, 0, stream>>>(vout[p], b_gcn + p * HH, NH, HH);
        spmm_scatter<<<(int)(((long)EE * 64 + 255) / 256), 256, 0, stream>>>(
            src + (long)p * EE, dst + (long)p * EE, vals + (long)p * EE, ft, vout[p]);
        prelu_store<<<(int)((NH + 255) / 256), 256, 0, stream>>>(vout[p], vB[p], a_gcn + p, NH);
    }

    // ---- student predictor ----
    wmma_gemm_bf16<<<dim3(PH / 32, gy), 256, 0, stream>>>(
        vB[0], W1B, b1, nullptr, xbuf, stats, nullptr, PH, HH, 0, MTILES);
    bn_prelu<<<(int)((NPH + 255) / 256), 256, 0, stream>>>(xbuf, stats, bn_g, bn_b, a_pred, NPH);
    wmma_gemm_bf16<<<dim3(HH / 32, gy), 256, 0, stream>>>(
        xbuf, W2B, b2, vpred, nullptr, nullptr, nullptr, HH, PH, 0, MTILES);

    // ---- distillation loss (teachers = views 1,2) ----
    loss_kernel<<<NN / 8, 256, 0, stream>>>(vpred, vout[1], vout[2], lossp);

    // ---- type-level attention: z order = [t1, t2, student] ----
    const int zorder[3] = {1, 2, 0};
    for (int zi = 0; zi < 3; ++zi) {
        wmma_gemm_bf16<<<dim3(HH / 32, gy), 256, 0, stream>>>(
            vB[zorder[zi]], WattB, b_att, nullptr, nullptr, nullptr, csum + zi * HH,
            HH, HH, 0, MTILES);
    }
    beta_kernel<<<1, 32, 0, stream>>>(csum, att, beta);
    combine_kernel<<<(int)((NH + 255) / 256), 256, 0, stream>>>(vout[1], vout[2], vout[0],
                                                               beta, zout, NH);
}
